// CausalMatrixGameSelfAttention_71390946394750
// MI455X (gfx1250) — matrix operand order repaired and checked by reading, not compiled, not run on hardware
//
#include <hip/hip_runtime.h>
#include <hip/hip_bf16.h>
#include <math.h>

#define N_HEADS   12
#define HEAD_DIM  128
#define KBLK      32
#define QTILE     16
#define WAVES     4
#define THREADS   (WAVES * 32)
#define LOCAL_ATTN 15
#define SINK_FRAMES 1

typedef __attribute__((ext_vector_type(16))) __bf16 v16bf;
typedef __attribute__((ext_vector_type(8)))  float  v8f;
typedef __attribute__((ext_vector_type(4)))  unsigned int v4u;
typedef __attribute__((ext_vector_type(8)))  int    v8i;
typedef __attribute__((ext_vector_type(4)))  int    v4i;

struct U16x16 { uint4 a, b; };        // 32B container -> v16bf
struct W8     { unsigned int w[8]; }; // 32B container -> v16bf

static __device__ inline unsigned short f2bf(float f) {
  union { float f; unsigned u; } x; x.f = f;
  unsigned r = x.u + 0x7FFFu + ((x.u >> 16) & 1u);   // round-to-nearest-even
  return (unsigned short)(r >> 16);
}

static __device__ inline v16bf make_v16bf(const unsigned int* w) {
  W8 t;
#pragma unroll
  for (int i = 0; i < 8; ++i) t.w[i] = w[i];
  return __builtin_bit_cast(v16bf, t);
}

// fragment: halfs[0..7] from p0, halfs[8..15] from p1 (b128 loads; LDS or global)
static __device__ inline v16bf frag16(const unsigned short* p0, const unsigned short* p1) {
  U16x16 t;
  t.a = *(const uint4*)p0;
  t.b = *(const uint4*)p1;
  return __builtin_bit_cast(v16bf, t);
}

// RoPE angle for complex-pair index c (head_dim=128 -> 64 pairs: 22 frame / 21 h / 21 w)
static __device__ inline void rope_cs(int c, int tok, int fs, int gw, int start_frame,
                                      float* cs, float* sn) {
  int f_idx = tok / fs;
  int rem   = tok - f_idx * fs;
  int hh    = rem / gw;
  int ww    = rem - hh * gw;
  float pos, expo;
  if (c < 22)      { pos = (float)(start_frame + f_idx); expo = (float)(2 * c)        * (1.0f / 44.0f); }
  else if (c < 43) { pos = (float)hh;                    expo = (float)(2 * (c - 22)) * (1.0f / 42.0f); }
  else             { pos = (float)ww;                    expo = (float)(2 * (c - 43)) * (1.0f / 42.0f); }
  float inv = __expf(-expo * 9.2103403719761836f);  // ln(10000)
  __sincosf(pos * inv, sn, cs);
}

struct WP { int kv_start, local_start, sink, evict, Ktot, start_frame, fs, gw; };

static __device__ inline WP window_params(int current_start, int global_end, int local_end_index,
                                          int gh, int gw, int L, int cache_len) {
  WP p;
  p.gw = gw;
  int fs = gh * gw; p.fs = fs;
  p.start_frame = current_start / fs;
  p.sink = SINK_FRAMES * fs;
  int max_attn = LOCAL_ATTN * fs;
  int current_end = current_start + L;
  int local_end;
  if (current_end > global_end && L + local_end_index > cache_len) {
    int n_evict = L + local_end_index - cache_len;
    p.evict   = n_evict;
    local_end = local_end_index + current_end - global_end - n_evict;
  } else {
    p.evict   = 0;
    local_end = local_end_index + current_end - global_end;
  }
  p.local_start = local_end - L;
  p.kv_start    = max(0, local_end - max_attn);
  p.Ktot        = local_end - p.kv_start;
  return p;
}

static __device__ inline v8f zero8() {
  v8f z = {0.f, 0.f, 0.f, 0.f, 0.f, 0.f, 0.f, 0.f};
  return z;
}

// ---------------- TDM descriptor builders (CDNA5 D#, 2-D tile) ----------------
static __device__ inline v4u tdm_group0(unsigned lds_addr, unsigned long long gaddr) {
  v4u g;
  g[0] = 1u;                                              // count=1, user descriptor
  g[1] = lds_addr;                                        // LDS byte address
  g[2] = (unsigned)(gaddr & 0xFFFFFFFFu);                 // global_addr[31:0]
  g[3] = (unsigned)((gaddr >> 32) & 0x1FFFFFFu) | (2u << 30);  // addr[56:32] | type=2
  return g;
}

// data_size=2B; pad enabled; 2-D tile: tile0 x tile1, row stride stride0 (elements)
static __device__ inline v8i tdm_group1(int pad_interval, int pad_amount,
                                        unsigned td0, unsigned td1,
                                        unsigned tile0, unsigned tile1,
                                        unsigned long long stride0) {
  v8i g;
  unsigned w0 = (1u << 16)                      // data_size = 2 bytes
              | (1u << 20)                      // pad_enable
              | ((unsigned)pad_interval << 22)
              | ((unsigned)pad_amount   << 25);
  g[0] = (int)w0;
  g[1] = (int)((td0 & 0xFFFFu) << 16);                         // tensor_dim0[15:0]
  g[2] = (int)((td0 >> 16) | ((td1 & 0xFFFFu) << 16));         // td0[31:16] | td1[15:0]
  g[3] = (int)((td1 >> 16) | (tile0 << 16));                   // td1[31:16] | tile_dim0
  g[4] = (int)(tile1 & 0xFFFFu);                               // tile_dim1 (tile_dim2=0)
  g[5] = (int)(stride0 & 0xFFFFFFFFu);                         // dim0 stride lo
  g[6] = (int)((stride0 >> 32) & 0xFFFFu);                     // dim0 stride hi (dim1 stride=0)
  g[7] = 0;
  return g;
}

static __device__ inline void tdm_load_2d(unsigned lds_addr, const unsigned short* gptr,
                                          int pad_interval, int pad_amount,
                                          unsigned td0, unsigned td1,
                                          unsigned tile0, unsigned tile1,
                                          unsigned long long stride0) {
  v4u g0 = tdm_group0(lds_addr, (unsigned long long)(uintptr_t)gptr);
  v8i g1 = tdm_group1(pad_interval, pad_amount, td0, td1, tile0, tile1, stride0);
  v4i z4 = {0, 0, 0, 0};
  v8i z8 = {0, 0, 0, 0, 0, 0, 0, 0};
  __builtin_amdgcn_tensor_load_to_lds(g0, g1, z4, z4, z8, 0);   // 6-arg form (clang-23)
}

// =======================  PHASE 1: pack K/V window  =========================
// Kb[h][key][d] = bf16(rope-or-cache K);  Vb[h][d][key] = bf16(V) (transposed)
__global__ __launch_bounds__(256)
void prep_kv_kernel(const float* __restrict__ k, const float* __restrict__ v,
                    const float* __restrict__ k_cache, const float* __restrict__ v_cache,
                    const int* cs_p, const int* ge_p, const int* le_p,
                    const int* gh_p, const int* gw_p,
                    unsigned short* __restrict__ Kb, unsigned short* __restrict__ Vb,
                    int L, int cache_len) {
  const WP wp = window_params(cs_p[0], ge_p[0], le_p[0], gh_p[0], gw_p[0], L, cache_len);
  const int h   = blockIdx.y;
  const int idx = blockIdx.x * 256 + threadIdx.x;   // (key << 6) | pair
  const int key = idx >> 6;
  const int dp  = idx & 63;
  if (key >= wp.Ktot) return;

  int ci = wp.kv_start + key;
  float2 kv, vv;
  if (ci >= wp.local_start) {                 // fresh token: RoPE(k), v
    int t = ci - wp.local_start;
    size_t off = ((size_t)t * N_HEADS + h) * HEAD_DIM + 2 * dp;
    kv = *(const float2*)(k + off);
    vv = *(const float2*)(v + off);
    float cs, sn;
    rope_cs(dp, t, wp.fs, wp.gw, wp.start_frame, &cs, &sn);
    float r0 = kv.x * cs - kv.y * sn;
    float r1 = kv.x * sn + kv.y * cs;
    kv.x = r0; kv.y = r1;
  } else {                                    // cache (virtually rolled past sink)
    int srci = ci + ((wp.evict && ci >= wp.sink) ? wp.evict : 0);
    size_t off = ((size_t)srci * N_HEADS + h) * HEAD_DIM + 2 * dp;
    kv = *(const float2*)(k_cache + off);
    vv = *(const float2*)(v_cache + off);
  }
  size_t kb_off = ((size_t)h * cache_len + key) * HEAD_DIM + 2 * dp;
  *(unsigned int*)&Kb[kb_off] = (unsigned)f2bf(kv.x) | ((unsigned)f2bf(kv.y) << 16);
  size_t vb_base = (size_t)h * cache_len * HEAD_DIM;
  Vb[vb_base + (size_t)(2 * dp)     * wp.Ktot + key] = f2bf(vv.x);
  Vb[vb_base + (size_t)(2 * dp + 1) * wp.Ktot + key] = f2bf(vv.y);
}

// Qb[h][q][d] = bf16(rope(q))
__global__ __launch_bounds__(256)
void prep_q_kernel(const float* __restrict__ q,
                   const int* cs_p, const int* gh_p, const int* gw_p,
                   unsigned short* __restrict__ Qb, int L) {
  const int h   = blockIdx.y;
  const int idx = blockIdx.x * 256 + threadIdx.x;
  const int qr  = idx >> 6;
  const int dp  = idx & 63;
  if (qr >= L) return;
  int fs = gh_p[0] * gw_p[0];
  int start_frame = cs_p[0] / fs;
  float2 t = *(const float2*)(q + ((size_t)qr * N_HEADS + h) * HEAD_DIM + 2 * dp);
  float cs, sn;
  rope_cs(dp, qr, fs, gw_p[0], start_frame, &cs, &sn);
  float r0 = t.x * cs - t.y * sn;
  float r1 = t.x * sn + t.y * cs;
  *(unsigned int*)&Qb[((size_t)h * L + qr) * HEAD_DIM + 2 * dp] =
      (unsigned)f2bf(r0) | ((unsigned)f2bf(r1) << 16);
}

// =======================  PHASE 2: TDM flash attention  =====================
__global__ __launch_bounds__(THREADS)
void attn_tdm_kernel(const unsigned short* __restrict__ Kb,
                     const unsigned short* __restrict__ Vb,
                     const unsigned short* __restrict__ Qb,
                     const int* cs_p, const int* ge_p, const int* le_p,
                     const int* gh_p, const int* gw_p,
                     float* __restrict__ out, int L, int cache_len) {
  __shared__ __align__(16) unsigned short Ks[2][KBLK][HEAD_DIM + 8];  // 272B rows
  __shared__ __align__(16) unsigned short Vt[2][HEAD_DIM][KBLK + 8];  // 80B rows

  const int tid  = threadIdx.x;
  const int lane = tid & 31;
  const int wave = tid >> 5;
  const int h    = blockIdx.x;
  const int q0   = (blockIdx.y * WAVES + wave) * QTILE;

  // uniform scalar params (SGPR-resident for TDM descriptors)
  const int cs_s = __builtin_amdgcn_readfirstlane(cs_p[0]);
  const int ge_s = __builtin_amdgcn_readfirstlane(ge_p[0]);
  const int le_s = __builtin_amdgcn_readfirstlane(le_p[0]);
  const int gh_s = __builtin_amdgcn_readfirstlane(gh_p[0]);
  const int gw_s = __builtin_amdgcn_readfirstlane(gw_p[0]);
  const WP wp = window_params(cs_s, ge_s, le_s, gh_s, gw_s, L, cache_len);

  const int   lm    = lane & 15;
  const int   lh    = lane >> 4;
  const float scale = 0.088388347648318447f;   // 1/sqrt(128)

  const unsigned short* Kb_h = Kb + (size_t)h * cache_len * HEAD_DIM;
  const unsigned short* Vb_h = Vb + (size_t)h * cache_len * HEAD_DIM;

  // ---- Q tile fragments (pre-roped bf16 from workspace) ----
  v16bf qb[4];
  {
    int qrow = q0 + lm;
    int tok  = (qrow < L) ? qrow : 0;
    const unsigned short* src = Qb + ((size_t)h * L + tok) * HEAD_DIM;
#pragma unroll
    for (int ds = 0; ds < 4; ++ds) {
      const unsigned short* p = src + ds * 32 + lh * 8;
      qb[ds] = frag16(p, p + 16);
    }
  }

  v8f o[8];
#pragma unroll
  for (int dt = 0; dt < 8; ++dt) o[dt] = zero8();
  float m_run = -INFINITY, l_run = 0.f;

  const int nblk = (wp.Ktot + KBLK - 1) / KBLK;

  // prologue: TDM-stage block 0 into buffer 0 (wave 0 only; EXEC-independent per wave)
  if (wave == 0) {
    tdm_load_2d((unsigned)(uintptr_t)&Ks[0][0][0], Kb_h,
                /*pi=*/5, /*pa=*/3, HEAD_DIM, (unsigned)wp.Ktot, HEAD_DIM, KBLK, HEAD_DIM);
    tdm_load_2d((unsigned)(uintptr_t)&Vt[0][0][0], Vb_h,
                /*pi=*/3, /*pa=*/3, (unsigned)wp.Ktot, HEAD_DIM, KBLK, HEAD_DIM,
                (unsigned long long)wp.Ktot);
  }

  for (int blk = 0; blk < nblk; ++blk) {
    const int kbase = blk * KBLK;
    const int buf   = blk & 1;

    if (wave == 0) {
      if (blk + 1 < nblk) {     // prefetch next block into the other buffer
        int nb = kbase + KBLK;
        tdm_load_2d((unsigned)(uintptr_t)&Ks[buf ^ 1][0][0],
                    Kb_h + (size_t)nb * HEAD_DIM,
                    5, 3, HEAD_DIM, (unsigned)(wp.Ktot - nb), HEAD_DIM, KBLK, HEAD_DIM);
        tdm_load_2d((unsigned)(uintptr_t)&Vt[buf ^ 1][0][0],
                    Vb_h + nb,
                    3, 3, (unsigned)(wp.Ktot - nb), HEAD_DIM, KBLK, HEAD_DIM,
                    (unsigned long long)wp.Ktot);
        __builtin_amdgcn_s_wait_tensorcnt(2);   // current block's pair complete
      } else {
        __builtin_amdgcn_s_wait_tensorcnt(0);
      }
    }
    __syncthreads();

    // ---- S^T = K * Q^T : two 16(key)x16(q) tiles over d=128 ----
    v8f c0 = zero8(), c1 = zero8();
    {
      const unsigned short* row0 = &Ks[buf][lm][0];
      const unsigned short* row1 = &Ks[buf][16 + lm][0];
      const int kb = lh * 8;
#pragma unroll
      for (int ds = 0; ds < 4; ++ds) {
        v16bf a0 = frag16(row0 + ds * 32 + kb, row0 + ds * 32 + 16 + kb);
        v16bf a1 = frag16(row1 + ds * 32 + kb, row1 + ds * 32 + 16 + kb);
        c0 = __builtin_amdgcn_wmma_f32_16x16x32_bf16(false, a0, false, qb[ds], (short)0, c0, false, false);
        c1 = __builtin_amdgcn_wmma_f32_16x16x32_bf16(false, a1, false, qb[ds], (short)0, c1, false, false);
      }
    }

    // ---- online softmax per query column ----
    float s0[8], s1[8], bm = -INFINITY;
#pragma unroll
    for (int r = 0; r < 8; ++r) {
      int k0 = kbase + lh * 8 + r;
      s0[r] = (k0      < wp.Ktot) ? c0[r] * scale : -INFINITY;
      s1[r] = (k0 + 16 < wp.Ktot) ? c1[r] * scale : -INFINITY;
      bm = fmaxf(bm, fmaxf(s0[r], s1[r]));
    }
    bm = fmaxf(bm, __shfl_xor(bm, 16));
    float nm    = fmaxf(m_run, bm);
    float alpha = __expf(m_run - nm);
    float rs = 0.f, p0[8], p1[8];
#pragma unroll
    for (int r = 0; r < 8; ++r) {
      p0[r] = __expf(s0[r] - nm);
      p1[r] = __expf(s1[r] - nm);
      rs += p0[r] + p1[r];
    }
    rs += __shfl_xor(rs, 16);
    l_run = l_run * alpha + rs;
    m_run = nm;
#pragma unroll
    for (int dt = 0; dt < 8; ++dt)
#pragma unroll
      for (int r = 0; r < 8; ++r) o[dt][r] *= alpha;

    // ---- P^T B-operand: fix 8/8 key split across lane halves ----
    float bl[8], bh[8];
#pragma unroll
    for (int i = 0; i < 8; ++i) {
      float sp0 = __shfl_xor(p0[i], 16);
      float sp1 = __shfl_xor(p1[i], 16);
      bl[i] = (lane < 16) ? p0[i] : sp1;
      bh[i] = (lane < 16) ? sp0 : p1[i];
    }
    unsigned int pw[8];
#pragma unroll
    for (int j = 0; j < 4; ++j) {
      pw[j]     = (unsigned)f2bf(bl[2 * j]) | ((unsigned)f2bf(bl[2 * j + 1]) << 16);
      pw[4 + j] = (unsigned)f2bf(bh[2 * j]) | ((unsigned)f2bf(bh[2 * j + 1]) << 16);
    }
    v16bf pb = make_v16bf(pw);

    // ---- O^T += V^T * P^T ----
    {
      const int kb = lh * 16;
#pragma unroll
      for (int dt = 0; dt < 8; ++dt) {
        const unsigned short* vr = &Vt[buf][dt * 16 + lm][0];
        v16bf va = frag16(vr + kb, vr + kb + 8);
        o[dt] = __builtin_amdgcn_wmma_f32_16x16x32_bf16(false, va, false, pb, (short)0, o[dt], false, false);
      }
    }
    __syncthreads();
  }

  int qrow = q0 + lm;
  if (qrow < L) {
    float invl = 1.0f / l_run;
#pragma unroll
    for (int dt = 0; dt < 8; ++dt) {
      float* dst = out + ((size_t)qrow * N_HEADS + h) * HEAD_DIM + dt * 16 + lh * 8;
      *(float4*)dst       = make_float4(o[dt][0] * invl, o[dt][1] * invl,
                                        o[dt][2] * invl, o[dt][3] * invl);
      *(float4*)(dst + 4) = make_float4(o[dt][4] * invl, o[dt][5] * invl,
                                        o[dt][6] * invl, o[dt][7] * invl);
    }
  }
}

// ================= Fallback: single-kernel version (round-1) ========
__global__ __launch_bounds__(THREADS)
void flash_attn_rope_kernel(const float* __restrict__ q,  const float* __restrict__ k,
                            const float* __restrict__ v,  const float* __restrict__ k_cache,
                            const float* __restrict__ v_cache,
                            const int* cs_p, const int* ge_p, const int* le_p,
                            const int* gh_p, const int* gw_p,
                            float* __restrict__ out, int L, int cache_len) {
  __shared__ __align__(16) unsigned short Ks[KBLK][HEAD_DIM + 8];
  __shared__ __align__(16) unsigned short Vt[HEAD_DIM][KBLK + 8];

  const int tid  = threadIdx.x;
  const int lane = tid & 31;
  const int wave = tid >> 5;
  const int h    = blockIdx.x;
  const int q0   = (blockIdx.y * WAVES + wave) * QTILE;

  const WP wp = window_params(cs_p[0], ge_p[0], le_p[0], gh_p[0], gw_p[0], L, cache_len);

  const int   lm    = lane & 15;
  const int   lh    = lane >> 4;
  const float scale = 0.088388347648318447f;

  v16bf qb[4];
  {
    int  qrow = q0 + lm;
    bool qv   = qrow < L;
    int  tok  = qv ? qrow : 0;
#pragma unroll
    for (int ds = 0; ds < 4; ++ds) {
      const float* src = q + ((size_t)tok * N_HEADS + h) * HEAD_DIM + ds * 32 + lh * 16;
      unsigned int w[8];
#pragma unroll
      for (int i = 0; i < 8; ++i) {
        float x0 = 0.f, x1 = 0.f;
        if (qv) { float2 t = *(const float2*)(src + 2 * i); x0 = t.x; x1 = t.y; }
        float cs, sn;
        rope_cs(ds * 16 + lh * 8 + i, tok, wp.fs, wp.gw, wp.start_frame, &cs, &sn);
        w[i] = (unsigned)f2bf(x0 * cs - x1 * sn) | ((unsigned)f2bf(x0 * sn + x1 * cs) << 16);
      }
      qb[ds] = make_v16bf(w);
    }
  }

  v8f o[8];
#pragma unroll
  for (int dt = 0; dt < 8; ++dt) o[dt] = zero8();
  float m_run = -INFINITY, l_run = 0.f;

  const int nblk = (wp.Ktot + KBLK - 1) / KBLK;
  for (int blk = 0; blk < nblk; ++blk) {
    const int kbase = blk * KBLK;
#pragma unroll
    for (int p = 0; p < 16; ++p) {
      int idx = tid + p * THREADS;
      int kl  = idx >> 6;
      int dp  = idx & 63;
      int kg  = kbase + kl;
      float2 kv = make_float2(0.f, 0.f), vv = make_float2(0.f, 0.f);
      if (kg < wp.Ktot) {
        int ci = wp.kv_start + kg;
        if (ci >= wp.local_start) {
          int t = ci - wp.local_start;
          size_t off = ((size_t)t * N_HEADS + h) * HEAD_DIM + 2 * dp;
          kv = *(const float2*)(k + off);
          vv = *(const float2*)(v + off);
          float cs, sn;
          rope_cs(dp, t, wp.fs, wp.gw, wp.start_frame, &cs, &sn);
          float r0 = kv.x * cs - kv.y * sn;
          float r1 = kv.x * sn + kv.y * cs;
          kv.x = r0; kv.y = r1;
        } else {
          int srci = ci + ((wp.evict && ci >= wp.sink) ? wp.evict : 0);
          size_t off = ((size_t)srci * N_HEADS + h) * HEAD_DIM + 2 * dp;
          kv = *(const float2*)(k_cache + off);
          vv = *(const float2*)(v_cache + off);
        }
      }
      *(unsigned int*)&Ks[kl][2 * dp] = (unsigned)f2bf(kv.x) | ((unsigned)f2bf(kv.y) << 16);
      Vt[2 * dp][kl]     = f2bf(vv.x);
      Vt[2 * dp + 1][kl] = f2bf(vv.y);
    }
    __syncthreads();

    v8f c0 = zero8(), c1 = zero8();
    {
      const unsigned short* row0 = &Ks[lm][0];
      const unsigned short* row1 = &Ks[16 + lm][0];
      const int kb = lh * 16;
#pragma unroll
      for (int ds = 0; ds < 4; ++ds) {
        v16bf a0 = frag16(row0 + ds * 32 + kb, row0 + ds * 32 + kb + 8);
        v16bf a1 = frag16(row1 + ds * 32 + kb, row1 + ds * 32 + kb + 8);
        c0 = __builtin_amdgcn_wmma_f32_16x16x32_bf16(false, a0, false, qb[ds], (short)0, c0, false, false);
        c1 = __builtin_amdgcn_wmma_f32_16x16x32_bf16(false, a1, false, qb[ds], (short)0, c1, false, false);
      }
    }

    float s0[8], s1[8], bm = -INFINITY;
#pragma unroll
    for (int r = 0; r < 8; ++r) {
      int k0 = kbase + lh * 8 + r;
      s0[r] = (k0      < wp.Ktot) ? c0[r] * scale : -INFINITY;
      s1[r] = (k0 + 16 < wp.Ktot) ? c1[r] * scale : -INFINITY;
      bm = fmaxf(bm, fmaxf(s0[r], s1[r]));
    }
    bm = fmaxf(bm, __shfl_xor(bm, 16));
    float nm    = fmaxf(m_run, bm);
    float alpha = __expf(m_run - nm);
    float rs = 0.f, p0[8], p1[8];
#pragma unroll
    for (int r = 0; r < 8; ++r) {
      p0[r] = __expf(s0[r] - nm);
      p1[r] = __expf(s1[r] - nm);
      rs += p0[r] + p1[r];
    }
    rs += __shfl_xor(rs, 16);
    l_run = l_run * alpha + rs;
    m_run = nm;
#pragma unroll
    for (int dt = 0; dt < 8; ++dt)
#pragma unroll
      for (int r = 0; r < 8; ++r) o[dt][r] *= alpha;

    float bl[8], bh[8];
#pragma unroll
    for (int i = 0; i < 8; ++i) {
      float sp0 = __shfl_xor(p0[i], 16);
      float sp1 = __shfl_xor(p1[i], 16);
      bl[i] = (lane < 16) ? p0[i] : sp1;
      bh[i] = (lane < 16) ? sp0 : p1[i];
    }
    unsigned int pw[8];
#pragma unroll
    for (int j = 0; j < 4; ++j) {
      pw[j]     = (unsigned)f2bf(bl[2 * j]) | ((unsigned)f2bf(bl[2 * j + 1]) << 16);
      pw[4 + j] = (unsigned)f2bf(bh[2 * j]) | ((unsigned)f2bf(bh[2 * j + 1]) << 16);
    }
    v16bf pb = make_v16bf(pw);

    {
      const int kb = lh * 16;
#pragma unroll
      for (int dt = 0; dt < 8; ++dt) {
        const unsigned short* vr = &Vt[dt * 16 + lm][0];
        v16bf va = frag16(vr + kb, vr + kb + 8);
        o[dt] = __builtin_amdgcn_wmma_f32_16x16x32_bf16(false, va, false, pb, (short)0, o[dt], false, false);
      }
    }
    __syncthreads();
  }

  int qrow = q0 + lm;
  if (qrow < L) {
    float invl = 1.0f / l_run;
#pragma unroll
    for (int dt = 0; dt < 8; ++dt) {
      float* dst = out + ((size_t)qrow * N_HEADS + h) * HEAD_DIM + dt * 16 + lh * 8;
      *(float4*)dst       = make_float4(o[dt][0] * invl, o[dt][1] * invl,
                                        o[dt][2] * invl, o[dt][3] * invl);
      *(float4*)(dst + 4) = make_float4(o[dt][4] * invl, o[dt][5] * invl,
                                        o[dt][6] * invl, o[dt][7] * invl);
    }
  }
}

extern "C" void kernel_launch(void* const* d_in, const int* in_sizes, int n_in,
                              void* d_out, int out_size, void* d_ws, size_t ws_size,
                              hipStream_t stream) {
  (void)n_in; (void)out_size;
  const float* q  = (const float*)d_in[0];
  const float* k  = (const float*)d_in[1];
  const float* v  = (const float*)d_in[2];
  const float* kc = (const float*)d_in[3];
  const float* vc = (const float*)d_in[4];
  const int* cs = (const int*)d_in[5];
  const int* ge = (const int*)d_in[6];
  const int* le = (const int*)d_in[7];
  // d_in[8] = grid_f (implied), unused
  const int* gh = (const int*)d_in[9];
  const int* gw = (const int*)d_in[10];

  const int L         = in_sizes[0] / (N_HEADS * HEAD_DIM);
  const int cache_len = in_sizes[3] / (N_HEADS * HEAD_DIM);

  const size_t kv_elems = (size_t)N_HEADS * cache_len * HEAD_DIM;   // per buffer
  const size_t q_elems  = (size_t)N_HEADS * L * HEAD_DIM;
  const size_t need     = (2 * kv_elems + q_elems) * sizeof(unsigned short);

  dim3 agrid(N_HEADS, (L + QTILE * WAVES - 1) / (QTILE * WAVES));

  if (ws_size >= need) {
    unsigned short* Kb = (unsigned short*)d_ws;
    unsigned short* Vb = Kb + kv_elems;
    unsigned short* Qb = Vb + kv_elems;

    dim3 pk_grid((unsigned)(((size_t)cache_len * 64 + 255) / 256), N_HEADS);
    prep_kv_kernel<<<pk_grid, 256, 0, stream>>>(k, v, kc, vc, cs, ge, le, gh, gw,
                                                Kb, Vb, L, cache_len);
    dim3 pq_grid((unsigned)(((size_t)L * 64 + 255) / 256), N_HEADS);
    prep_q_kernel<<<pq_grid, 256, 0, stream>>>(q, cs, gh, gw, Qb, L);

    attn_tdm_kernel<<<agrid, THREADS, 0, stream>>>(Kb, Vb, Qb, cs, ge, le, gh, gw,
                                                   (float*)d_out, L, cache_len);
  } else {
    flash_attn_rope_kernel<<<agrid, THREADS, 0, stream>>>(
        q, k, v, kc, vc, cs, ge, le, gh, gw, (float*)d_out, L, cache_len);
  }
}
